// FCOS_30760555774283
// MI455X (gfx1250) — matrix operand order repaired and checked by reading, not compiled, not run on hardware
//
#include <hip/hip_runtime.h>
#include <hip/hip_bf16.h>

// ---------------------------------------------------------------------------
// FCOS head on gfx1250: implicit-GEMM conv3x3 via v_wmma_f32_16x16x32_bf16,
// TDM (tensor_load_to_lds) staging of weight slabs, LDS-sourced B fragments.
// ---------------------------------------------------------------------------

typedef __attribute__((ext_vector_type(16))) __bf16 v16bf;
typedef __attribute__((ext_vector_type(8)))  __bf16 v8bf;
typedef __attribute__((ext_vector_type(8)))  float  v8f;
typedef __attribute__((ext_vector_type(4)))  unsigned int v4u;
typedef __attribute__((ext_vector_type(8)))  int v8i;
typedef __attribute__((ext_vector_type(4)))  int v4i;

#define CIN    256
#define NBATCH 4
#define PTOT   21824   // 128^2 + 64^2 + 32^2 + 16^2 + 8^2
#define NOUT   85      // 80 cls + 4 bbox + 1 ctr

#if __has_builtin(__builtin_amdgcn_tensor_load_to_lds) && \
    __has_builtin(__builtin_amdgcn_s_wait_tensorcnt)
#define HAVE_TDM 1
#else
#define HAVE_TDM 0
#endif

__device__ __forceinline__ __bf16 f2bf(float f) {
    unsigned int u = __builtin_bit_cast(unsigned int, f);
    unsigned int r = (u + 0x7FFFu + ((u >> 16) & 1u)) >> 16;   // RNE
    unsigned short s = (unsigned short)r;
    return __builtin_bit_cast(__bf16, s);
}

#if HAVE_TDM
// 1-D TDM copy: nelem 2-byte elements, global -> LDS (one issue per call).
// This toolchain's builtin takes 6 args: (g0 v4u, g1 v8i, g2 v4i, g3 v4i,
// g4 v8i, cpol i32).
__device__ __forceinline__ void tdm_load_1d(unsigned lds_off, const void* gsrc,
                                            unsigned nelem) {
    unsigned long long ga = (unsigned long long)gsrc;
    v4u g0;
    g0[0] = 1u;                                          // count=1, user mode
    g0[1] = lds_off;                                     // lds_addr
    g0[2] = (unsigned)(ga & 0xFFFFFFFFu);                // global_addr[31:0]
    g0[3] = (unsigned)(((ga >> 32) & 0x01FFFFFFu) | 0x80000000u); // [56:32]+type=2
    v8i g1;
    g1[0] = 0x00010000;                                  // wg_mask=0, data_size=2B
    g1[1] = (int)((nelem & 0xFFFFu) << 16);              // tensor_dim0 lo16
    g1[2] = (int)((nelem >> 16) & 0xFFFFu);              // tensor_dim0 hi16
    g1[3] = (int)((nelem & 0xFFFFu) << 16);              // tile_dim0
    g1[4] = 0;                                           // tile_dim1/2 unused
    g1[5] = (int)nelem;                                  // tensor_dim0_stride lo32
    g1[6] = 0;
    g1[7] = 0;
    v4i z4 = {0, 0, 0, 0};
    v8i z8 = {0, 0, 0, 0, 0, 0, 0, 0};
    __builtin_amdgcn_tensor_load_to_lds(g0, g1, z4, z4, z8, 0);
}
#endif

// ---- zero a small bf16 buffer ---------------------------------------------
__global__ __launch_bounds__(256)
void zero_bf16(__bf16* __restrict__ p, int n) {
    int i = blockIdx.x * 256 + threadIdx.x;
    if (i < n) p[i] = f2bf(0.0f);
}

// ---- NCHW fp32 -> NHWC bf16 ------------------------------------------------
__global__ __launch_bounds__(256)
void pack_input(const float* __restrict__ src, __bf16* __restrict__ dst,
                int H, int W) {
    const size_t HW  = (size_t)H * W;
    const size_t tot = (size_t)NBATCH * CIN * HW;
    size_t idx = (size_t)blockIdx.x * 256 + threadIdx.x;
    if (idx >= tot) return;
    size_t chw = (size_t)CIN * HW;
    int    n   = (int)(idx / chw);
    size_t r   = idx - (size_t)n * chw;
    int    c   = (int)(r / HW);
    size_t pix = r - (size_t)c * HW;
    dst[((size_t)n * HW + pix) * CIN + c] = f2bf(src[idx]);
}

// ---- OIHW fp32 -> [tap][Opad][cin] bf16, zero padded -----------------------
__global__ __launch_bounds__(256)
void pack_weights(const float* __restrict__ src, __bf16* __restrict__ dst,
                  int Opad, int Oreal) {
    int tot = 9 * Opad * CIN;
    int idx = blockIdx.x * 256 + threadIdx.x;
    if (idx >= tot) return;
    int t  = idx / (Opad * CIN);
    int r  = idx - t * (Opad * CIN);
    int o  = r / CIN;
    int ci = r - o * CIN;
    __bf16 v = f2bf(0.0f);
    if (o < Oreal) v = f2bf(src[((size_t)(o * CIN + ci)) * 9 + t]);
    dst[idx] = v;
}

// ---------------------------------------------------------------------------
// Trunk conv3x3 (256->256): one wave = 16 pixels x 64 out-channels (OT=4).
// Block = 8 waves covering 128 pixels; B slab (64 x 256 bf16 per tap, 32 KB)
// staged to LDS via TDM, double buffered. Grids are always exact (no partial
// blocks) so barriers are safe.
// ---------------------------------------------------------------------------
__global__ __launch_bounds__(256)
void conv3x3_trunk(const __bf16* __restrict__ X,
                   const __bf16* __restrict__ Wt,
                   const float* __restrict__ bias,
                   float* __restrict__ outF,
                   const __bf16* __restrict__ zrow,
                   int H, int W) {
    __shared__ __attribute__((aligned(64))) __bf16 sB[2][64 * CIN]; // 2 x 32 KB

    const int HW   = H * W;
    const int lane = threadIdx.x & 31;
    const int wave = threadIdx.x >> 5;
    const int half = lane >> 4;
    const int lm   = lane & 15;
    const int pb   = (blockIdx.x * 8 + wave) * 16;
    const int o0   = blockIdx.y * 64;

    // this lane's A-row pixel
    const int p  = pb + lm;
    const int n  = p / HW;
    const int rr = p - n * HW;
    const int h  = rr / W;
    const int w  = rr - h * W;

    auto stage = [&](int t, int buf) {
        const __bf16* src = Wt + ((size_t)t * CIN + o0) * CIN;
#if HAVE_TDM
        if (wave == 0)
            tdm_load_1d((unsigned)(unsigned long long)&sB[buf][0], src, 64 * CIN);
#else
        v8bf* d = (v8bf*)&sB[buf][0];
        const v8bf* s = (const v8bf*)src;
        for (int i = threadIdx.x; i < (64 * CIN) / 8; i += 256) d[i] = s[i];
#endif
    };

    v8f acc[4] = {};

    stage(0, 0);
#if HAVE_TDM
    if (wave == 0) __builtin_amdgcn_s_wait_tensorcnt(0);
#endif
    __syncthreads();

    for (int t = 0; t < 9; ++t) {
        const int cur = t & 1;
        if (t < 8) stage(t + 1, cur ^ 1);          // prefetch next tap's slab

        const int dy = t / 3 - 1;
        const int dx = t - (t / 3) * 3 - 1;
        const int y = h + dy, x = w + dx;
        const bool av = (y >= 0) & (y < H) & (x >= 0) & (x < W);
        const __bf16* arow = av ? (X + ((size_t)(n * H + y) * W + x) * CIN) : zrow;

#pragma unroll
        for (int kc = 0; kc < 8; ++kc) {
            union { v16bf v; v8bf h8[2]; } ua;
            ua.h8[0] = *(const v8bf*)(arow + kc * 32 + half * 16);
            ua.h8[1] = *(const v8bf*)(arow + kc * 32 + half * 16 + 8);
#pragma unroll
            for (int ot = 0; ot < 4; ++ot) {
                v16bf b = *(const v16bf*)(&sB[cur][(ot * 16 + lm) * CIN +
                                                   kc * 32 + half * 16]);
                acc[ot] = __builtin_amdgcn_wmma_f32_16x16x32_bf16(
                              false, ua.v, false, b, (short)0, acc[ot],
                              false, false);
            }
        }
#if HAVE_TDM
        if (wave == 0 && t < 8) __builtin_amdgcn_s_wait_tensorcnt(0);
#endif
        __syncthreads();
    }

#pragma unroll
    for (int ot = 0; ot < 4; ++ot) {
        const int   oc = o0 + ot * 16 + lm;
        const float bv = bias[oc];
#pragma unroll
        for (int r8 = 0; r8 < 8; ++r8) {
            int pp = pb + r8 + half * 8;
            outF[(size_t)pp * CIN + oc] = acc[ot][r8] + bv;
        }
    }
}

// ---------------------------------------------------------------------------
// Final conv3x3 (256 -> Opad<=80): one wave = 16 pixels x 16 out-channels,
// direct global B loads, writes into d_out[n, level_off+hw, 85].
// ---------------------------------------------------------------------------
__global__ __launch_bounds__(256)
void conv3x3_final(const __bf16* __restrict__ X,
                   const __bf16* __restrict__ Wt,
                   const float* __restrict__ bias,
                   float* __restrict__ outD,
                   const __bf16* __restrict__ zrow,
                   int H, int W, int Opad, int Oreal,
                   int level_off, int ch_off, int relu) {
    const int HW   = H * W;
    const int lane = threadIdx.x & 31;
    const int wave = threadIdx.x >> 5;
    const int half = lane >> 4;
    const int lm   = lane & 15;
    const int pb   = (blockIdx.x * 8 + wave) * 16;
    const int o0   = blockIdx.y * 16;

    const int p  = pb + lm;
    const int n  = p / HW;
    const int rr = p - n * HW;
    const int h  = rr / W;
    const int w  = rr - h * W;

    const __bf16* brow = Wt + ((size_t)(o0 + lm)) * CIN + (size_t)half * 16;

    v8f acc = {};
    for (int t = 0; t < 9; ++t) {
        const int dy = t / 3 - 1;
        const int dx = t - (t / 3) * 3 - 1;
        const int y = h + dy, x = w + dx;
        const bool av = (y >= 0) & (y < H) & (x >= 0) & (x < W);
        const __bf16* arow = av ? (X + ((size_t)(n * H + y) * W + x) * CIN) : zrow;
        const __bf16* bt   = brow + (size_t)t * Opad * CIN;
#pragma unroll
        for (int kc = 0; kc < 8; ++kc) {
            union { v16bf v; v8bf h8[2]; } ua;
            ua.h8[0] = *(const v8bf*)(arow + kc * 32 + half * 16);
            ua.h8[1] = *(const v8bf*)(arow + kc * 32 + half * 16 + 8);
            v16bf b = *(const v16bf*)(bt + kc * 32);
            acc = __builtin_amdgcn_wmma_f32_16x16x32_bf16(
                      false, ua.v, false, b, (short)0, acc, false, false);
        }
    }

    const int oc = o0 + lm;
    if (oc < Oreal) {
        const float bv = bias[oc];
#pragma unroll
        for (int r8 = 0; r8 < 8; ++r8) {
            int pp = pb + r8 + half * 8;
            float v = acc[r8] + bv;
            if (relu) v = v > 0.0f ? v : 0.0f;
            int nn   = pp / HW;
            int rpos = pp - nn * HW;
            outD[((size_t)nn * PTOT + level_off + rpos) * NOUT + ch_off + oc] = v;
        }
    }
}

// ---- GroupNorm stats: one block per (n, group) -----------------------------
__global__ __launch_bounds__(256)
void gn_stats(const float* __restrict__ F, float* __restrict__ stats, int HW) {
    const int n   = blockIdx.x >> 4;
    const int g   = blockIdx.x & 15;
    const int tid = threadIdx.x;
    const int total = HW * 16;
    float s = 0.0f, ss = 0.0f;
    for (int i = tid; i < total; i += 256) {
        int   pix = i >> 4;
        int   ch  = (g << 4) + (i & 15);
        float v   = F[((size_t)n * HW + pix) * CIN + ch];
        s += v;  ss += v * v;
    }
    __shared__ float sh1[256], sh2[256];
    sh1[tid] = s; sh2[tid] = ss;
    __syncthreads();
    for (int st = 128; st > 0; st >>= 1) {
        if (tid < st) { sh1[tid] += sh1[tid + st]; sh2[tid] += sh2[tid + st]; }
        __syncthreads();
    }
    if (tid == 0) {
        float mean = sh1[0] / total;
        float var  = sh2[0] / total - mean * mean;
        stats[blockIdx.x * 2]     = mean;
        stats[blockIdx.x * 2 + 1] = rsqrtf(var + 1e-5f);
    }
}

// ---- GroupNorm apply + ReLU, fp32 NHWC -> bf16 NHWC ------------------------
__global__ __launch_bounds__(256)
void gn_apply_relu(const float* __restrict__ F, const float* __restrict__ stats,
                   const float* __restrict__ gamma, const float* __restrict__ beta,
                   __bf16* __restrict__ out, int HW) {
    const size_t tot = (size_t)NBATCH * HW * CIN;
    size_t idx = (size_t)blockIdx.x * 256 + threadIdx.x;
    if (idx >= tot) return;
    int    ch  = (int)(idx & (CIN - 1));
    size_t pix = idx >> 8;
    int    n   = (int)(pix / HW);
    int    g   = ch >> 4;
    float mean = stats[(n * 16 + g) * 2];
    float rstd = stats[(n * 16 + g) * 2 + 1];
    float v = (F[idx] - mean) * rstd * gamma[ch] + beta[ch];
    v = v > 0.0f ? v : 0.0f;
    out[idx] = f2bf(v);
}

// ---------------------------------------------------------------------------
extern "C" void kernel_launch(void* const* d_in, const int* in_sizes, int n_in,
                              void* d_out, int out_size, void* d_ws, size_t ws_size,
                              hipStream_t stream) {
    (void)in_sizes; (void)n_in; (void)out_size; (void)ws_size;

    // jax pytree flatten order (sorted dict keys):
    // 0-4 feats; 5 bbox_b; 6 bbox_w; 7 cls_logits_b; 8 cls_logits_w;
    // 9-12 cls_trunk[0]{b,beta,gamma,w}; 13-16 cls_trunk[1]{...};
    // 17 ctr_b; 18 ctr_w; 19-22 reg_trunk[0]; 23-26 reg_trunk[1]
    const float* feat[5];
    for (int i = 0; i < 5; ++i) feat[i] = (const float*)d_in[i];
    const float* bbox_b = (const float*)d_in[5];
    const float* bbox_w = (const float*)d_in[6];
    const float* clsl_b = (const float*)d_in[7];
    const float* clsl_w = (const float*)d_in[8];
    struct TP { const float *b, *beta, *gamma, *w; };
    TP cls0{(const float*)d_in[9],  (const float*)d_in[10], (const float*)d_in[11], (const float*)d_in[12]};
    TP cls1{(const float*)d_in[13], (const float*)d_in[14], (const float*)d_in[15], (const float*)d_in[16]};
    const float* ctr_b = (const float*)d_in[17];
    const float* ctr_w = (const float*)d_in[18];
    TP reg0{(const float*)d_in[19], (const float*)d_in[20], (const float*)d_in[21], (const float*)d_in[22]};
    TP reg1{(const float*)d_in[23], (const float*)d_in[24], (const float*)d_in[25], (const float*)d_in[26]};

    // workspace carve-out
    const size_t maxElems = (size_t)NBATCH * CIN * 128 * 128;
    char* p = (char*)d_ws;
    __bf16* I16 = (__bf16*)p;  p += maxElems * 2;               // packed input
    __bf16* T16 = (__bf16*)p;  p += maxElems * 2;               // trunk temp
    float*  Fb  = (float*)p;   p += maxElems * 4;               // conv fp32 out
    __bf16* Wc0 = (__bf16*)p;  p += (size_t)9 * 256 * CIN * 2;
    __bf16* Wc1 = (__bf16*)p;  p += (size_t)9 * 256 * CIN * 2;
    __bf16* Wr0 = (__bf16*)p;  p += (size_t)9 * 256 * CIN * 2;
    __bf16* Wr1 = (__bf16*)p;  p += (size_t)9 * 256 * CIN * 2;
    __bf16* Wlg = (__bf16*)p;  p += (size_t)9 * 80  * CIN * 2;
    __bf16* Wbb = (__bf16*)p;  p += (size_t)9 * 16  * CIN * 2;
    __bf16* Wct = (__bf16*)p;  p += (size_t)9 * 16  * CIN * 2;
    float*  stats = (float*)p; p += 64 * 2 * sizeof(float);
    __bf16* zrow = (__bf16*)p;

    zero_bf16<<<1, 256, 0, stream>>>(zrow, CIN);

    auto packW = [&](const float* s, __bf16* d, int Opad, int Oreal) {
        int tot = 9 * Opad * CIN;
        pack_weights<<<(tot + 255) / 256, 256, 0, stream>>>(s, d, Opad, Oreal);
    };
    packW(cls0.w, Wc0, 256, 256);
    packW(cls1.w, Wc1, 256, 256);
    packW(reg0.w, Wr0, 256, 256);
    packW(reg1.w, Wr1, 256, 256);
    packW(clsl_w, Wlg, 80, 80);
    packW(bbox_w, Wbb, 16, 4);
    packW(ctr_w,  Wct, 16, 1);

    const int HWs[5][2] = {{128,128},{64,64},{32,32},{16,16},{8,8}};
    int lvl_off = 0;
    for (int L = 0; L < 5; ++L) {
        const int H = HWs[L][0], W = HWs[L][1], HW = H * W;
        size_t tot = (size_t)NBATCH * CIN * HW;
        pack_input<<<(unsigned)((tot + 255) / 256), 256, 0, stream>>>(feat[L], I16, H, W);

        const int Ptot   = NBATCH * HW;            // multiple of 128 at all levels
        const int blocksX = Ptot / 128;            // 8 waves x 16 pixels per block

        auto trunk = [&](const __bf16* Xb, const __bf16* Wb, const float* bb) {
            dim3 grid(blocksX, 4);                 // 4 x 64 out-channels
            conv3x3_trunk<<<grid, 256, 0, stream>>>(Xb, Wb, bb, Fb, zrow, H, W);
        };
        auto fin = [&](const __bf16* Xb, const __bf16* Wb, const float* bb,
                       int Opad, int Oreal, int ch_off, int relu) {
            dim3 grid(blocksX, Opad / 16);
            conv3x3_final<<<grid, 256, 0, stream>>>(Xb, Wb, bb, (float*)d_out,
                                                    zrow, H, W, Opad, Oreal,
                                                    lvl_off, ch_off, relu);
        };
        auto gn = [&](const TP& tp, __bf16* outb) {
            gn_stats<<<NBATCH * 16, 256, 0, stream>>>(Fb, stats, HW);
            size_t te = (size_t)NBATCH * HW * CIN;
            gn_apply_relu<<<(unsigned)((te + 255) / 256), 256, 0, stream>>>(
                Fb, stats, tp.gamma, tp.beta, outb, HW);
        };

        // classification head
        trunk(I16, Wc0, cls0.b);  gn(cls0, T16);
        trunk(T16, Wc1, cls1.b);  gn(cls1, T16);
        fin(T16, Wlg, clsl_b, 80, 80, 0, 0);       // cls logits -> ch 0..79

        // regression head (shared trunk for bbox + centerness)
        trunk(I16, Wr0, reg0.b);  gn(reg0, T16);
        trunk(T16, Wr1, reg1.b);  gn(reg1, T16);
        fin(T16, Wbb, bbox_b, 16, 4, 80, 1);       // bbox (ReLU) -> ch 80..83
        fin(T16, Wct, ctr_b,  16, 1, 84, 0);       // centerness  -> ch 84

        lvl_off += HW;
    }
}
